// RNN_3899830304968
// MI455X (gfx1250) — matrix operand order repaired and checked by reading, not compiled, not run on hardware
//
#include <hip/hip_runtime.h>

#define B_   128
#define T_   256
#define I_   512
#define H_   1024
#define NBLK 32        // persistent grid: (H/128) x (B/32) = 8 x 4

typedef __attribute__((ext_vector_type(16))) __bf16 v16bf;
typedef __attribute__((ext_vector_type(8)))  __bf16 v8bf;
typedef __attribute__((ext_vector_type(8)))  float  v8f;

union V16 { v16bf v; v8bf h[2]; unsigned short u[16]; };

__device__ __forceinline__ unsigned short f2bf(float f) {
  unsigned int u = __float_as_uint(f);
  u += 0x7FFFu + ((u >> 16) & 1u);           // round-to-nearest-even bf16
  return (unsigned short)(u >> 16);
}

// ---------------------------------------------------------------------------
// Per call: convert weights to bf16, fold biases, zero h0 + barrier counters.
// ---------------------------------------------------------------------------
__global__ void rnn_prep_kernel(const float* __restrict__ W_ih,
                                const float* __restrict__ W_hh,
                                const float* __restrict__ b_ih,
                                const float* __restrict__ b_hh,
                                unsigned short* __restrict__ wih_bf,
                                unsigned short* __restrict__ whh_bf,
                                float* __restrict__ bias,
                                unsigned short* __restrict__ h0,
                                unsigned int* __restrict__ bar) {
  const int i = blockIdx.x * blockDim.x + threadIdx.x;
  if (i < H_ * I_) wih_bf[i] = f2bf(W_ih[i]);
  if (i < H_ * H_) whh_bf[i] = f2bf(W_hh[i]);
  if (i < H_)      bias[i]   = b_ih[i] + b_hh[i];
  if (i < B_ * H_) h0[i]     = 0;
  if (i < T_)      bar[i]    = 0;
}

// ---------------------------------------------------------------------------
// Phase 1: xp = x @ W_ih^T + (b_ih + b_hh), written into d_out states region.
// Block = 8 waves arranged 4(M) x 2(N); wave tile 16x32; block tile 64x64.
// ---------------------------------------------------------------------------
__global__ __launch_bounds__(256) void xp_gemm_kernel(
    const float* __restrict__ x, const unsigned short* __restrict__ wih,
    const float* __restrict__ bias, float* __restrict__ out) {
  const int lane = threadIdx.x & 31;
  const int wv   = threadIdx.x >> 5;
  const int l16  = lane & 15;
  const int hi   = lane >> 4;

  const int m0 = blockIdx.y * 64 + (wv >> 1) * 16;   // row tile in [0, B*T)
  const int n0 = blockIdx.x * 64 + (wv & 1) * 32;    // col tile in [0, H)

  const float*          ap  = x   + (size_t)(m0 + l16) * I_ + hi * 16;
  const unsigned short* bp0 = wih + (size_t)(n0 + l16) * I_ + hi * 16;
  const unsigned short* bp1 = wih + (size_t)(n0 + 16 + l16) * I_ + hi * 16;

  const float bias0 = bias[n0 + l16];
  const float bias1 = bias[n0 + 16 + l16];
  v8f c0, c1;
  for (int v = 0; v < 8; ++v) { c0[v] = bias0; c1[v] = bias1; }

  for (int k0 = 0; k0 < I_; k0 += 32) {
    const float4 f0 = *(const float4*)(ap + k0);
    const float4 f1 = *(const float4*)(ap + k0 + 4);
    const float4 f2 = *(const float4*)(ap + k0 + 8);
    const float4 f3 = *(const float4*)(ap + k0 + 12);
    V16 a;
    a.u[0]  = f2bf(f0.x); a.u[1]  = f2bf(f0.y); a.u[2]  = f2bf(f0.z); a.u[3]  = f2bf(f0.w);
    a.u[4]  = f2bf(f1.x); a.u[5]  = f2bf(f1.y); a.u[6]  = f2bf(f1.z); a.u[7]  = f2bf(f1.w);
    a.u[8]  = f2bf(f2.x); a.u[9]  = f2bf(f2.y); a.u[10] = f2bf(f2.z); a.u[11] = f2bf(f2.w);
    a.u[12] = f2bf(f3.x); a.u[13] = f2bf(f3.y); a.u[14] = f2bf(f3.z); a.u[15] = f2bf(f3.w);

    V16 b0, b1;
    b0.v = *(const v16bf*)(bp0 + k0);
    b1.v = *(const v16bf*)(bp1 + k0);

    c0 = __builtin_amdgcn_wmma_f32_16x16x32_bf16(false, a.v, false, b0.v, (short)0, c0, false, false);
    c1 = __builtin_amdgcn_wmma_f32_16x16x32_bf16(false, a.v, false, b1.v, (short)0, c1, false, false);

    if (k0 + 32 < I_) __builtin_prefetch(ap + k0 + 32, 0, 0);
  }

  for (int v = 0; v < 8; ++v) {
    const size_t m = (size_t)(m0 + v + hi * 8);
    out[m * H_ + n0 + l16]      = c0[v];
    out[m * H_ + n0 + 16 + l16] = c1[v];
  }
}

// ---------------------------------------------------------------------------
// Phase 2: PERSISTENT kernel — all T timesteps, device-wide split barrier per
// step (fresh counter per t, release/acquire at AGENT scope).
// Grid = 32 blocks of 8 waves (2(M) x 4(N)); wave tile 16x32; block 32x128.
// ---------------------------------------------------------------------------
__global__ __launch_bounds__(256) void rnn_persistent_kernel(
    const unsigned short* __restrict__ whh,
    float* __restrict__ states,
    unsigned short* __restrict__ hbuf0,
    unsigned short* __restrict__ hbuf1,
    float* __restrict__ h_last,
    unsigned int* __restrict__ bar) {
  const int lane = threadIdx.x & 31;
  const int wv   = threadIdx.x >> 5;
  const int l16  = lane & 15;
  const int hi   = lane >> 4;

  const int m0 = blockIdx.y * 32 + (wv >> 2) * 16;   // batch-row tile
  const int n0 = blockIdx.x * 128 + (wv & 3) * 32;   // hidden-col tile

  // Loop-invariant W_hh fragment pointers (L2-resident, 2 MB).
  const unsigned short* bp0 = whh + (size_t)(n0 + l16) * H_ + hi * 16;
  const unsigned short* bp1 = whh + (size_t)(n0 + 16 + l16) * H_ + hi * 16;
  const size_t a_off = (size_t)(m0 + l16) * H_ + hi * 16;

  for (int t = 0; t < T_; ++t) {
    const unsigned short* h_in  = (t & 1) ? hbuf1 : hbuf0;
    unsigned short*       h_out = (t & 1) ? hbuf0 : hbuf1;
    const unsigned short* ap = h_in + a_off;

    // C init = xp slice for timestep t (states layout [B, T, H])
    v8f c0, c1;
    for (int v = 0; v < 8; ++v) {
      const size_t r = ((size_t)(m0 + v + hi * 8) * T_ + t) * H_;
      c0[v] = states[r + n0 + l16];
      c1[v] = states[r + n0 + 16 + l16];
    }

    #pragma unroll 4
    for (int k0 = 0; k0 < H_; k0 += 32) {
      V16 a;
      a.h[0] = *(const v8bf*)(ap + k0);        // K = k0 + hi*16 + 0..7
      a.h[1] = *(const v8bf*)(ap + k0 + 8);    // K = k0 + hi*16 + 8..15

      V16 b0, b1;
      b0.v = *(const v16bf*)(bp0 + k0);
      b1.v = *(const v16bf*)(bp1 + k0);

      c0 = __builtin_amdgcn_wmma_f32_16x16x32_bf16(false, a.v, false, b0.v, (short)0, c0, false, false);
      c1 = __builtin_amdgcn_wmma_f32_16x16x32_bf16(false, a.v, false, b1.v, (short)0, c1, false, false);
    }

    // Epilogue: tanh; final states are never re-read -> nontemporal stores.
    for (int v = 0; v < 8; ++v) {
      const int m    = m0 + v + hi * 8;
      const size_t r = ((size_t)m * T_ + t) * H_;
      const float y0 = tanhf(c0[v]);
      const float y1 = tanhf(c1[v]);
      __builtin_nontemporal_store(y0, &states[r + n0 + l16]);
      __builtin_nontemporal_store(y1, &states[r + n0 + 16 + l16]);
      h_out[(size_t)m * H_ + n0 + l16]      = f2bf(y0);   // read next step: keep RT
      h_out[(size_t)m * H_ + n0 + 16 + l16] = f2bf(y1);
      if (t == T_ - 1) {
        h_last[(size_t)m * H_ + n0 + l16]      = y0;
        h_last[(size_t)m * H_ + n0 + 16 + l16] = y1;
      }
    }

    // Device-wide split barrier: make h_out visible, then sync all 32 blocks.
    __threadfence();
    __syncthreads();
    if (threadIdx.x == 0) {
      __hip_atomic_fetch_add(&bar[t], 1u, __ATOMIC_RELEASE, __HIP_MEMORY_SCOPE_AGENT);
      while (__hip_atomic_load(&bar[t], __ATOMIC_ACQUIRE, __HIP_MEMORY_SCOPE_AGENT) < NBLK) {
        __builtin_amdgcn_s_sleep(1);
      }
    }
    __syncthreads();
  }
}

extern "C" void kernel_launch(void* const* d_in, const int* in_sizes, int n_in,
                              void* d_out, int out_size, void* d_ws, size_t ws_size,
                              hipStream_t stream) {
  (void)in_sizes; (void)n_in; (void)out_size; (void)ws_size;
  const float* x    = (const float*)d_in[0];
  const float* W_ih = (const float*)d_in[1];
  const float* W_hh = (const float*)d_in[2];
  const float* b_ih = (const float*)d_in[3];
  const float* b_hh = (const float*)d_in[4];
  float* out = (float*)d_out;

  // Workspace layout (all offsets well-aligned): ~3.7 MB total
  char* ws = (char*)d_ws;
  unsigned short* wih_bf = (unsigned short*)(ws);                                   // 1 MB
  unsigned short* whh_bf = (unsigned short*)(ws + (size_t)H_ * I_ * 2);             // 2 MB
  float*          biasws = (float*)(ws + (size_t)H_ * I_ * 2 + (size_t)H_ * H_ * 2);// 4 KB
  unsigned short* hbuf0  = (unsigned short*)((char*)biasws + (size_t)H_ * 4);       // 256 KB
  unsigned short* hbuf1  = hbuf0 + (size_t)B_ * H_;                                 // 256 KB
  unsigned int*   bar    = (unsigned int*)(hbuf1 + (size_t)B_ * H_);                // 1 KB

  rnn_prep_kernel<<<(H_ * H_ + 255) / 256, 256, 0, stream>>>(
      W_ih, W_hh, b_ih, b_hh, wih_bf, whh_bf, biasws, hbuf0, bar);

  xp_gemm_kernel<<<dim3(H_ / 64, (B_ * T_) / 64), 256, 0, stream>>>(
      x, wih_bf, biasws, out);

  rnn_persistent_kernel<<<dim3(H_ / 128, B_ / 32), 256, 0, stream>>>(
      whh_bf, out, hbuf0, hbuf1, out + (size_t)B_ * T_ * H_, bar);
}
